// GNN_4569845203242
// MI455X (gfx1250) — matrix operand order repaired and checked by reading, not compiled, not run on hardware
//
#include <hip/hip_runtime.h>
#include <hip/hip_bf16.h>

#define DH 256  // hidden width (D_HID)

typedef __attribute__((ext_vector_type(16))) __bf16 v16bf;
typedef __attribute__((ext_vector_type(8)))  __bf16 v8bf;
typedef __attribute__((ext_vector_type(8)))  float  v8f;

union BF16x16 { v16bf v; v8bf h[2]; };

// Guaranteed hardware no-return FP atomic (L2 atomic unit), STOREcnt-tracked.
static __device__ __forceinline__ void atomic_add_f32(float* p, float v) {
  asm volatile("global_atomic_add_f32 %0, %1, off" : : "v"(p), "v"(v) : "memory");
}

// A-matrix fragment (16x32 bf16, ISA 7.12.2): lane l -> row = l&15, half h = l>>4,
// elements K = [8h..8h+7] then [16+8h..16+8h+7] (two 16B contiguous chunks).
static __device__ __forceinline__ v16bf load_a_frag(const __bf16* __restrict__ base,
                                                    int ld, int r0, int k0, int lane) {
  int m = lane & 15, hh = lane >> 4;
  const __bf16* p = base + (size_t)(r0 + m) * ld + k0 + hh * 8;
  BF16x16 u;
  u.h[0] = *(const v8bf*)(p);
  u.h[1] = *(const v8bf*)(p + 16);
  return u.v;
}

// B-matrix fragment (32x16 bf16): lane l -> col n = l&15, half h = l>>4,
// elements K = [8h..8h+7] then [16+8h..16+8h+7] (two 16B chunks). B[k][n] = W[n][k] since
// W is stored [out,in] row-major -> read straight from W's row n.
static __device__ __forceinline__ v16bf load_b_frag(const __bf16* __restrict__ base,
                                                    int ld, int c0, int k0, int lane) {
  int n = lane & 15, hh = lane >> 4;
  const __bf16* p = base + (size_t)(c0 + n) * ld + k0 + hh * 8;
  BF16x16 u;
  u.h[0] = *(const v8bf*)(p);
  u.h[1] = *(const v8bf*)(p + 16);
  return u.v;
}

// T[N x 256] = H[N x K] * W^T, bf16 in / f32 accumulate / bf16 out.
// Block = 256 threads = 8 waves; wave (rs,cg) owns rows [rbase..rbase+15],
// cols [cg*64 .. cg*64+63] as 4 WMMA accumulators; A frag amortized over 4 B frags.
__global__ __launch_bounds__(256) void k_gemm_bf16(const __bf16* __restrict__ H,
                                                   const __bf16* __restrict__ Wb,
                                                   __bf16* __restrict__ T,
                                                   int N, int K) {
  int lane  = threadIdx.x & 31;
  int wave  = threadIdx.x >> 5;
  int rbase = blockIdx.x * 32 + (wave >> 2) * 16;
  int nbase = (wave & 3) * 64;
  if (rbase >= N) return;  // wave-uniform: EXEC stays all-ones for WMMA
  v8f a0 = {}, a1 = {}, a2 = {}, a3 = {};
  for (int k0 = 0; k0 < K; k0 += 32) {
    v16bf a  = load_a_frag(H,  K, rbase,      k0, lane);
    v16bf b0 = load_b_frag(Wb, K, nbase +  0, k0, lane);
    v16bf b1 = load_b_frag(Wb, K, nbase + 16, k0, lane);
    v16bf b2 = load_b_frag(Wb, K, nbase + 32, k0, lane);
    v16bf b3 = load_b_frag(Wb, K, nbase + 48, k0, lane);
    a0 = __builtin_amdgcn_wmma_f32_16x16x32_bf16(false, a, false, b0, (short)0, a0, false, false);
    a1 = __builtin_amdgcn_wmma_f32_16x16x32_bf16(false, a, false, b1, (short)0, a1, false, false);
    a2 = __builtin_amdgcn_wmma_f32_16x16x32_bf16(false, a, false, b2, (short)0, a2, false, false);
    a3 = __builtin_amdgcn_wmma_f32_16x16x32_bf16(false, a, false, b3, (short)0, a3, false, false);
  }
  // D layout (ISA 7.12.2): lane l -> col n = l&15; VGPR r -> row r + 8*(l>>4).
  int n = lane & 15, hh = lane >> 4;
#pragma unroll
  for (int r = 0; r < 8; ++r) {
    __bf16* o = T + (size_t)(rbase + hh * 8 + r) * DH + nbase + n;
    o[0]  = (__bf16)a0[r];
    o[16] = (__bf16)a1[r];
    o[32] = (__bf16)a2[r];
    o[48] = (__bf16)a3[r];
  }
}

__global__ void k_fill_f32(float* __restrict__ p, float v, int n) {
  int i = blockIdx.x * blockDim.x + threadIdx.x;
  if (i < n) p[i] = v;
}

__global__ void k_deg(const int* __restrict__ col, float* __restrict__ deg, int E) {
  int e = blockIdx.x * blockDim.x + threadIdx.x;
  if (e < E) atomic_add_f32(&deg[col[e]], 1.0f);
}

__global__ void k_rsqrt(float* __restrict__ d, int n) {
  int i = blockIdx.x * blockDim.x + threadIdx.x;
  if (i < n) d[i] = rsqrtf(d[i]);  // deg >= 1 always (self-loop)
}

__global__ void k_norm(const float* __restrict__ dinv, const int* __restrict__ row,
                       const int* __restrict__ col, float* __restrict__ norm, int E) {
  int e = blockIdx.x * blockDim.x + threadIdx.x;
  if (e < E) norm[e] = dinv[row[e]] * dinv[col[e]];
}

__global__ void k_cvt_bf16(const float* __restrict__ s, __bf16* __restrict__ d, int n) {
  int i = blockIdx.x * blockDim.x + threadIdx.x;
  if (i < n) d[i] = (__bf16)s[i];
}

// One wave per edge: lane gathers 8 bf16 feats (one b128), 8 f32 L2 atomics per lane.
// Edge metadata (row/col/norm) forced wave-uniform -> SMEM scalar loads via K$.
// Working set (bf16 t 51MB + f32 acc 102MB) fits MI455X 192MB L2 -> no HBM thrash.
__global__ __launch_bounds__(256) void k_scatter(const __bf16* __restrict__ t,
                                                 const int* __restrict__ row,
                                                 const int* __restrict__ col,
                                                 const float* __restrict__ norm,
                                                 float* __restrict__ acc, int E) {
  int tid  = blockIdx.x * blockDim.x + threadIdx.x;
  int lane = tid & 31;
  int e    = __builtin_amdgcn_readfirstlane(tid >> 5);  // wave-uniform edge id
  if (e >= E) return;                                   // wave-uniform branch
  int   r = row[e];     // uniform index -> scalar (SMEM) loads
  int   c = col[e];
  float w = norm[e];
  v8bf  m = *(const v8bf*)(t + (size_t)r * DH + lane * 8);
  float* dst = acc + (size_t)c * DH + lane * 8;
#pragma unroll
  for (int i = 0; i < 8; ++i)
    atomic_add_f32(dst + i, (float)m[i] * w);
}

// h_next = bf16(relu(acc + self_loop)), self_loop = t[i]*dinv[i]^2 (no atomics needed)
__global__ void k_finish_relu(const float* __restrict__ acc, const __bf16* __restrict__ t,
                              const float* __restrict__ dinv, __bf16* __restrict__ h,
                              int total) {
  int i = blockIdx.x * blockDim.x + threadIdx.x;
  if (i >= total) return;
  float di = dinv[i >> 8];
  float v  = acc[i] + (float)t[i] * di * di;
  h[i] = (__bf16)(v > 0.f ? v : 0.f);
}

__global__ void k_finish_out(float* __restrict__ out, const __bf16* __restrict__ t,
                             const float* __restrict__ dinv, int total) {
  int i = blockIdx.x * blockDim.x + threadIdx.x;
  if (i >= total) return;
  float di = dinv[i >> 8];
  out[i] += (float)t[i] * di * di;
}

extern "C" void kernel_launch(void* const* d_in, const int* in_sizes, int n_in,
                              void* d_out, int out_size, void* d_ws, size_t ws_size,
                              hipStream_t stream) {
  (void)n_in; (void)out_size; (void)ws_size;
  const float* x  = (const float*)d_in[0];
  const int*   ei = (const int*)d_in[1];   // [2, E] row-major
  const float* W1 = (const float*)d_in[2]; // [256,128]
  const float* W2 = (const float*)d_in[3]; // [256,256]
  const float* W3 = (const float*)d_in[4]; // [256,256]

  const int DIN = 128;
  const int E = in_sizes[1] / 2;
  const int N = in_sizes[0] / DIN;
  const int* row = ei;        // sources
  const int* col = ei + E;    // targets (aggregation)

  auto al = [](size_t v) { return (v + 255) & ~(size_t)255; };
  char* w = (char*)d_ws;
  float*  dinv = (float*)w;  w += al((size_t)N * 4);
  float*  norm = (float*)w;  w += al((size_t)E * 4);
  __bf16* hbf  = (__bf16*)w; w += al((size_t)N * DH * 2);
  __bf16* tbf  = (__bf16*)w; w += al((size_t)N * DH * 2);
  float*  acc  = (float*)w;  w += al((size_t)N * DH * 4);
  __bf16* wbf  = (__bf16*)w; w += al((size_t)DH * DH * 2);

  const int B = 256;
  auto gup = [](long long n, int b) { return (unsigned)((n + b - 1) / b); };
  const int total = N * DH;

  // ---- normalization (computed once, reused by all 3 layers) ----
  k_fill_f32<<<gup(N, B), B, 0, stream>>>(dinv, 1.0f, N);            // self-loop
  k_deg<<<gup(E, B), B, 0, stream>>>(col, dinv, E);
  k_rsqrt<<<gup(N, B), B, 0, stream>>>(dinv, N);
  k_norm<<<gup(E, B), B, 0, stream>>>(dinv, row, col, norm, E);
  k_cvt_bf16<<<gup((long long)N * DIN, B), B, 0, stream>>>(x, hbf, N * DIN);

  // ---- layer 1: 128 -> 256, relu ----
  k_cvt_bf16<<<gup(DH * DIN, B), B, 0, stream>>>(W1, wbf, DH * DIN);
  k_gemm_bf16<<<gup(N, 32), B, 0, stream>>>(hbf, wbf, tbf, N, DIN);
  k_fill_f32<<<gup(total, B), B, 0, stream>>>(acc, 0.f, total);
  k_scatter<<<gup((long long)E * 32, B), B, 0, stream>>>(tbf, row, col, norm, acc, E);
  k_finish_relu<<<gup(total, B), B, 0, stream>>>(acc, tbf, dinv, hbf, total);

  // ---- layer 2: 256 -> 256, relu ----
  k_cvt_bf16<<<gup(DH * DH, B), B, 0, stream>>>(W2, wbf, DH * DH);
  k_gemm_bf16<<<gup(N, 32), B, 0, stream>>>(hbf, wbf, tbf, N, DH);
  k_fill_f32<<<gup(total, B), B, 0, stream>>>(acc, 0.f, total);
  k_scatter<<<gup((long long)E * 32, B), B, 0, stream>>>(tbf, row, col, norm, acc, E);
  k_finish_relu<<<gup(total, B), B, 0, stream>>>(acc, tbf, dinv, hbf, total);

  // ---- layer 3: 256 -> 256, accumulate directly into d_out (f32) ----
  float* out = (float*)d_out;
  k_cvt_bf16<<<gup(DH * DH, B), B, 0, stream>>>(W3, wbf, DH * DH);
  k_gemm_bf16<<<gup(N, 32), B, 0, stream>>>(hbf, wbf, tbf, N, DH);
  k_fill_f32<<<gup(total, B), B, 0, stream>>>(out, 0.f, total);
  k_scatter<<<gup((long long)E * 32, B), B, 0, stream>>>(tbf, row, col, norm, out, E);
  k_finish_out<<<gup(total, B), B, 0, stream>>>(out, tbf, dinv, total);
}
